// AttentionGraphFusion_46961172414853
// MI455X (gfx1250) — matrix operand order repaired and checked by reading, not compiled, not run on hardware
//
#include <hip/hip_runtime.h>
#include <hip/hip_bf16.h>

#define Bsz 32
#define Ssz 8192
#define Dsz 512
#define Nn  32
#define Hh  8
#define HDsz 64
#define ODim 512

typedef __attribute__((ext_vector_type(16))) __bf16 v16bf;
typedef __attribute__((ext_vector_type(8)))  float  v8f;

__device__ __forceinline__ float sigm(float x) { return 1.0f / (1.0f + __expf(-x)); }

// A-fragment for v_wmma_f32_16x16x32_bf16: lane holds 16 contiguous bf16 at p
// (K = kb..kb+15).
__device__ __forceinline__ v16bf load_bf_frag(const __bf16* p) {
  v16bf o;
#pragma unroll
  for (int e = 0; e < 8; ++e) { o[e] = p[e]; o[8 + e] = p[8 + e]; }
  return o;
}

// ---------------------------------------------------------------- converts
__global__ __launch_bounds__(256)
void cvt_bf16_kernel(const float* __restrict__ src, unsigned short* __restrict__ dst_, int n) {
  __bf16* dst = (__bf16*)dst_;
  int i = blockIdx.x * 256 + threadIdx.x;
  if (i < n) dst[i] = (__bf16)src[i];
}

// W_e1 is [512,1024]; split into W1a = W_e1[:, :512], W1b = W_e1[:, 512:]
__global__ __launch_bounds__(256)
void split_e1_kernel(const float* __restrict__ src, unsigned short* __restrict__ a_,
                     unsigned short* __restrict__ b_) {
  __bf16* a = (__bf16*)a_;
  __bf16* b = (__bf16*)b_;
  int i = blockIdx.x * 256 + threadIdx.x;
  if (i < Dsz * Dsz) {
    int r = i >> 9, c = i & 511;
    a[i] = (__bf16)src[(size_t)r * 1024 + c];
    b[i] = (__bf16)src[(size_t)r * 1024 + 512 + c];
  }
}

// ---------------------------------------------------------------- node scores
// scores[b,s] = sigmoid( relu(x[b,s]@W_nc1^T + b_nc1) . w_nc2 + b_nc2 )
//
// The entire 256KB bf16 W_nc1 is staged once per workgroup into LDS with the
// gfx1250 async global->LDS DMA path (global_load_async_to_lds_b128, tracked
// by ASYNCcnt), then each wave runs 8 token tiles of 16x16x32 bf16 WMMA with
// B-fragments served from LDS (ds_load_b128) and only the x stream on VMEM.
__global__ __launch_bounds__(256)
void score_kernel(const float* __restrict__ x, const unsigned short* __restrict__ Wnc1_,
                  const float* __restrict__ bnc1, const float* __restrict__ Wnc2,
                  const float* __restrict__ bnc2, float* __restrict__ scores) {
  extern __shared__ char smem_raw[];            // 262144 bytes at launch
  __bf16* smem = (__bf16*)smem_raw;

  // ---- async stage: 256 threads x 64 iters x 16B = 256KB ----
#pragma unroll 4
  for (int c = 0; c < 64; ++c) {
    unsigned off = (unsigned)(c * 4096 + threadIdx.x * 16);
    const void* gp = (const char*)Wnc1_ + off;
    asm volatile("global_load_async_to_lds_b128 %0, %1, off"
                 :: "v"(off), "v"(gp) : "memory");
  }
  asm volatile("s_wait_asynccnt 0x0" ::: "memory");
  __syncthreads();

  const int lane  = threadIdx.x & 31;
  const int r     = lane & 15;
  const int khalf = lane >> 4;
  const int n     = r;
  const int kq16  = khalf * 16;
  const int wid   = blockIdx.x * 8 + (threadIdx.x >> 5);   // 0..2047
  const float b2  = bnc2[0];

#pragma unroll 1
  for (int itT = 0; itT < 8; ++itT) {
    const int tile = wid + itT * 2048;         // 0..16383 (512 tiles per batch)
    const int bIdx = tile >> 9;
    const int s0   = (tile & 511) * 16;
    const float* xr = x + ((size_t)bIdx * Ssz + s0 + r) * Dsz;

    // Preload all 16 A-fragments (the full 16x512 token tile, read once).
    v16bf afrag[16];
#pragma unroll
    for (int kt = 0; kt < 16; ++kt) {
      const int kb = kt * 32 + khalf * 16;
      const float4 q0 = *(const float4*)(xr + kb);
      const float4 q1 = *(const float4*)(xr + kb + 4);
      const float4 q2 = *(const float4*)(xr + kb + 8);
      const float4 q3 = *(const float4*)(xr + kb + 12);
      v16bf a;
      a[0]  = (__bf16)q0.x; a[1]  = (__bf16)q0.y; a[2]  = (__bf16)q0.z; a[3]  = (__bf16)q0.w;
      a[4]  = (__bf16)q1.x; a[5]  = (__bf16)q1.y; a[6]  = (__bf16)q1.z; a[7]  = (__bf16)q1.w;
      a[8]  = (__bf16)q2.x; a[9]  = (__bf16)q2.y; a[10] = (__bf16)q2.z; a[11] = (__bf16)q2.w;
      a[12] = (__bf16)q3.x; a[13] = (__bf16)q3.y; a[14] = (__bf16)q3.z; a[15] = (__bf16)q3.w;
      afrag[kt] = a;
    }

    float dot[8];
#pragma unroll
    for (int v = 0; v < 8; ++v) dot[v] = 0.f;

#pragma unroll 1
    for (int nt = 0; nt < 16; ++nt) {
      const __bf16* wrow = smem + (size_t)(nt * 16 + n) * Dsz + kq16;
      v8f acc = {};
#pragma unroll
      for (int kt = 0; kt < 16; ++kt) {
        v16bf bf = *(const v16bf*)(wrow + kt * 32);   // ds_load_b128 x2
        acc = __builtin_amdgcn_wmma_f32_16x16x32_bf16(false, afrag[kt], false, bf,
                                                      (short)0, acc, false, false);
      }
      const int col   = nt * 16 + n;
      const float bias = bnc1[col];
      const float w2   = Wnc2[col];
#pragma unroll
      for (int v = 0; v < 8; ++v) {
        float h = fmaxf(acc[v] + bias, 0.f);
        dot[v] = fmaf(h, w2, dot[v]);
      }
    }

    // Reduce across the 16 lanes of each half (C/D layout: lanes 0-15 rows v,
    // lanes 16-31 rows v+8), then sigmoid and store.
#pragma unroll
    for (int v = 0; v < 8; ++v) {
      float s = dot[v];
#pragma unroll
      for (int m = 1; m < 16; m <<= 1) s += __shfl_xor(s, m, 32);
      if (n == 0) scores[(size_t)bIdx * Ssz + s0 + khalf * 8 + v] = sigm(s + b2);
    }
  }
}

// ---------------------------------------------------------------- top-k + gather
__global__ __launch_bounds__(256)
void topk_kernel(const float* __restrict__ scores, float* __restrict__ scratch,
                 const float* __restrict__ x, int* __restrict__ topidx,
                 float* __restrict__ nodesf, unsigned short* __restrict__ nodesb_) {
  __bf16* nodesb = (__bf16*)nodesb_;
  const int b = blockIdx.x;
  const int tid = threadIdx.x;
  const float* sc = scores + (size_t)b * Ssz;
  float* scr = scratch + (size_t)b * Ssz;
  for (int i = tid; i < Ssz; i += 256) scr[i] = sc[i];
  __syncthreads();
  __shared__ float smax[256];
  __shared__ int   simax[256];
  __shared__ int   ssel;
  for (int it = 0; it < Nn; ++it) {
    float best = -1e30f; int bi = Ssz;
    for (int i = tid; i < Ssz; i += 256) {
      float v = scr[i];
      if (v > best || (v == best && i < bi)) { best = v; bi = i; }
    }
    smax[tid] = best; simax[tid] = bi;
    __syncthreads();
    for (int off = 128; off > 0; off >>= 1) {
      if (tid < off) {
        float ov = smax[tid + off]; int oi = simax[tid + off];
        if (ov > smax[tid] || (ov == smax[tid] && oi < simax[tid])) {
          smax[tid] = ov; simax[tid] = oi;
        }
      }
      __syncthreads();
    }
    if (tid == 0) { ssel = simax[0]; topidx[b * Nn + it] = ssel; scr[simax[0]] = -1e30f; }
    __syncthreads();
    const int sel = ssel;
    const float* xr = x + ((size_t)b * Ssz + sel) * Dsz;
    float*  nf = nodesf + ((size_t)b * Nn + it) * Dsz;
    __bf16* nb = nodesb + ((size_t)b * Nn + it) * Dsz;
    for (int c = tid; c < Dsz; c += 256) { float v = xr[c]; nf[c] = v; nb[c] = (__bf16)v; }
    __syncthreads();
  }
}

// ---------------------------------------------------------------- U/V/Q/K GEMMs
// Out[b] = nodes[b](32x512) @ W^T(512x512) (+bias for Q/K). One wave per 16x16
// output tile; 16 chained wmma along K.
__global__ __launch_bounds__(256)
void proj_kernel(const unsigned short* __restrict__ nodesb_,
                 const unsigned short* __restrict__ W1a_, const unsigned short* __restrict__ W1b_,
                 const unsigned short* __restrict__ Wq_,  const unsigned short* __restrict__ Wk_,
                 const float* __restrict__ bq, const float* __restrict__ bk,
                 float* __restrict__ U, float* __restrict__ V,
                 float* __restrict__ Q, float* __restrict__ Ko) {
  const __bf16* nodesb = (const __bf16*)nodesb_;
  const int wave = threadIdx.x >> 5, lane = threadIdx.x & 31;
  int t = blockIdx.x;
  const int blk   = t & 7; t >>= 3;
  const int which = t & 3; t >>= 2;
  const int b     = t;
  const int tile  = blk * 8 + wave;   // 0..63
  const int mtile = tile & 1;
  const int nt    = tile >> 1;        // 0..31
  const __bf16* W = (which == 0) ? (const __bf16*)W1a_
                  : (which == 1) ? (const __bf16*)W1b_
                  : (which == 2) ? (const __bf16*)Wq_ : (const __bf16*)Wk_;
  float* Out = (which == 0) ? U : (which == 1) ? V : (which == 2) ? Q : Ko;
  const int r = lane & 15, khalf = lane >> 4, n = r, kq16 = khalf * 16;
  const __bf16* arow = nodesb + ((size_t)b * Nn + mtile * 16 + r) * Dsz;
  const __bf16* wrow = W + (size_t)(nt * 16 + n) * Dsz + kq16;
  v8f acc = {};
#pragma unroll
  for (int kt = 0; kt < 16; ++kt) {
    v16bf a  = load_bf_frag(arow + kt * 32 + khalf * 16);
    v16bf bf = *(const v16bf*)(wrow + kt * 32);
    acc = __builtin_amdgcn_wmma_f32_16x16x32_bf16(false, a, false, bf,
                                                  (short)0, acc, false, false);
  }
  const float bias = (which == 2) ? bq[nt * 16 + n] : (which == 3) ? bk[nt * 16 + n] : 0.f;
  float* orow = Out + ((size_t)b * Nn + mtile * 16 + khalf * 8) * Dsz + nt * 16 + n;
#pragma unroll
  for (int v = 0; v < 8; ++v) orow[(size_t)v * Dsz] = acc[v] + bias;
}

// ---------------------------------------------------------------- edge weights
// edge[b,i,j] = edge[b,j,i] = sigmoid( sum_c relu(U[b,i,c]+V[b,j,c]+b_e1[c])*w_e2[c] + b_e2 )
// for i<j; diagonal = 0. One wave per (i,j).
__global__ __launch_bounds__(256)
void edge_kernel(const float* __restrict__ U, const float* __restrict__ V,
                 const float* __restrict__ be1, const float* __restrict__ We2,
                 const float* __restrict__ be2, float* __restrict__ edge) {
  const int gw   = (blockIdx.x * 256 + threadIdx.x) >> 5;
  const int lane = threadIdx.x & 31;
  const int b = gw >> 10;
  const int p = gw & 1023;
  const int i = p >> 5, j = p & 31;
  if (i == j) { if (lane == 0) edge[(size_t)b * Nn * Nn + i * Nn + j] = 0.f; return; }
  if (i > j) return;
  const float* u = U + ((size_t)b * Nn + i) * Dsz;
  const float* v = V + ((size_t)b * Nn + j) * Dsz;
  float s = 0.f;
  for (int c = lane; c < Dsz; c += 32) {
    float h = fmaxf(u[c] + v[c] + be1[c], 0.f);
    s = fmaf(h, We2[c], s);
  }
#pragma unroll
  for (int m = 1; m < 32; m <<= 1) s += __shfl_xor(s, m, 32);
  if (lane == 0) {
    float e = sigm(s + be2[0]);
    edge[(size_t)b * Nn * Nn + i * Nn + j] = e;
    edge[(size_t)b * Nn * Nn + j * Nn + i] = e;
  }
}

// ---------------------------------------------------------------- attention + fuse + out
__global__ __launch_bounds__(256)
void final_kernel(const float* __restrict__ Q, const float* __restrict__ K,
                  const float* __restrict__ edge, const float* __restrict__ nodes,
                  const float* __restrict__ Wout, const float* __restrict__ bout,
                  float* __restrict__ out) {
  __shared__ float logits[Hh][Nn][Nn];
  __shared__ float attn[Nn][Nn];
  __shared__ float wmat[Nn][Nn];
  __shared__ float colw[Nn];
  __shared__ float grep[Dsz];
  const int b = blockIdx.x;
  const int tid = threadIdx.x;
  const float* Qb = Q + (size_t)b * Nn * Dsz;
  const float* Kb = K + (size_t)b * Nn * Dsz;
  // logits[h,i,j] = q_i^h . k_j^h / 8
  for (int p = tid; p < Nn * Nn; p += 256) {
    int i = p >> 5, j = p & 31;
    const float* qi = Qb + (size_t)i * Dsz;
    const float* kj = Kb + (size_t)j * Dsz;
#pragma unroll
    for (int h = 0; h < Hh; ++h) {
      float s = 0.f;
      const float* qh = qi + h * HDsz;
      const float* kh = kj + h * HDsz;
      for (int c = 0; c < HDsz; ++c) s = fmaf(qh[c], kh[c], s);
      logits[h][i][j] = s * 0.125f;
    }
  }
  __syncthreads();
  { // per-(h,i) row softmax: exactly 256 rows
    int h = tid >> 5, i = tid & 31;
    float mx = -1e30f;
    for (int j = 0; j < Nn; ++j) mx = fmaxf(mx, logits[h][i][j]);
    float sum = 0.f;
    for (int j = 0; j < Nn; ++j) { float e = __expf(logits[h][i][j] - mx); logits[h][i][j] = e; sum += e; }
    float inv = 1.0f / sum;
    for (int j = 0; j < Nn; ++j) logits[h][i][j] *= inv;
  }
  __syncthreads();
  for (int p = tid; p < Nn * Nn; p += 256) {
    int i = p >> 5, j = p & 31;
    float s = 0.f;
#pragma unroll
    for (int h = 0; h < Hh; ++h) s += logits[h][i][j];
    attn[i][j] = s * (1.0f / Hh);
  }
  __syncthreads();
  if (tid < Nn) { // w = softmax(attn * edge_w) over j
    int i = tid;
    float vals[Nn]; float mx = -1e30f;
    for (int j = 0; j < Nn; ++j) {
      float v = attn[i][j] * edge[(size_t)b * Nn * Nn + i * Nn + j];
      vals[j] = v; mx = fmaxf(mx, v);
    }
    float sum = 0.f;
    for (int j = 0; j < Nn; ++j) { float e = __expf(vals[j] - mx); vals[j] = e; sum += e; }
    float inv = 1.f / sum;
    for (int j = 0; j < Nn; ++j) wmat[i][j] = vals[j] * inv;
  }
  __syncthreads();
  if (tid < Nn) { // column sums: graph_rep = (1/N) * (sum_i w[i,:]) @ nodes
    int k = tid; float s = 0.f;
    for (int i = 0; i < Nn; ++i) s += wmat[i][k];
    colw[k] = s;
  }
  __syncthreads();
  for (int c = tid; c < Dsz; c += 256) {
    float s = 0.f;
    for (int k = 0; k < Nn; ++k) s = fmaf(colw[k], nodes[((size_t)b * Nn + k) * Dsz + c], s);
    grep[c] = s * (1.0f / Nn);
  }
  __syncthreads();
  for (int o = tid; o < ODim; o += 256) {
    const float* wr = Wout + (size_t)o * Dsz;
    float s = bout[o];
    for (int c = 0; c < Dsz; ++c) s = fmaf(grep[c], wr[c], s);
    out[(size_t)b * ODim + o] = s;
  }
}

// ---------------------------------------------------------------- launcher
extern "C" void kernel_launch(void* const* d_in, const int* in_sizes, int n_in,
                              void* d_out, int out_size, void* d_ws, size_t ws_size,
                              hipStream_t stream) {
  (void)in_sizes; (void)n_in; (void)out_size; (void)ws_size;
  const float* x    = (const float*)d_in[0];
  const float* Wnc1 = (const float*)d_in[1];
  const float* bnc1 = (const float*)d_in[2];
  const float* Wnc2 = (const float*)d_in[3];
  const float* bnc2 = (const float*)d_in[4];
  const float* Wqf  = (const float*)d_in[5];
  const float* bq   = (const float*)d_in[6];
  const float* Wkf  = (const float*)d_in[7];
  const float* bk   = (const float*)d_in[8];
  const float* We1  = (const float*)d_in[9];
  const float* be1  = (const float*)d_in[10];
  const float* We2  = (const float*)d_in[11];
  const float* be2  = (const float*)d_in[12];
  const float* Wout = (const float*)d_in[13];
  const float* bout = (const float*)d_in[14];
  float* out = (float*)d_out;

  char* base = (char*)d_ws;
  size_t off = 0;
  auto alloc = [&](size_t bytes) -> char* {
    char* p = base + off;
    off = (off + bytes + 255) & ~(size_t)255;
    return p;
  };
  float*          scores  = (float*)alloc((size_t)Bsz * Ssz * 4);
  float*          scratch = (float*)alloc((size_t)Bsz * Ssz * 4);
  int*            topidx  = (int*)alloc((size_t)Bsz * Nn * 4);
  float*          nodesf  = (float*)alloc((size_t)Bsz * Nn * Dsz * 4);
  unsigned short* nodesb  = (unsigned short*)alloc((size_t)Bsz * Nn * Dsz * 2);
  unsigned short* Wnc1b   = (unsigned short*)alloc((size_t)256 * 512 * 2);
  unsigned short* W1a     = (unsigned short*)alloc((size_t)512 * 512 * 2);
  unsigned short* W1b     = (unsigned short*)alloc((size_t)512 * 512 * 2);
  unsigned short* Wqb     = (unsigned short*)alloc((size_t)512 * 512 * 2);
  unsigned short* Wkb     = (unsigned short*)alloc((size_t)512 * 512 * 2);
  float*          U       = (float*)alloc((size_t)Bsz * Nn * Dsz * 4);
  float*          V       = (float*)alloc((size_t)Bsz * Nn * Dsz * 4);
  float*          Qm      = (float*)alloc((size_t)Bsz * Nn * Dsz * 4);
  float*          Km      = (float*)alloc((size_t)Bsz * Nn * Dsz * 4);
  float*          edge    = (float*)alloc((size_t)Bsz * Nn * Nn * 4);

  cvt_bf16_kernel<<<(256 * 512 + 255) / 256, 256, 0, stream>>>(Wnc1, Wnc1b, 256 * 512);
  cvt_bf16_kernel<<<(512 * 512 + 255) / 256, 256, 0, stream>>>(Wqf, Wqb, 512 * 512);
  cvt_bf16_kernel<<<(512 * 512 + 255) / 256, 256, 0, stream>>>(Wkf, Wkb, 512 * 512);
  split_e1_kernel<<<(512 * 512 + 255) / 256, 256, 0, stream>>>(We1, W1a, W1b);

  // 256 blocks x 8 waves x 8 tile-iterations = all 16384 token tiles.
  // 256KB dynamic LDS per workgroup holds the full bf16 W_nc1 (ISA: up to
  // 320KB LDS per workgroup on CDNA5).
  score_kernel<<<256, 256, 262144, stream>>>(x, Wnc1b, bnc1, Wnc2, bnc2, scores);
  topk_kernel<<<Bsz, 256, 0, stream>>>(scores, scratch, x, topidx, nodesf, nodesb);
  proj_kernel<<<Bsz * 4 * 8, 256, 0, stream>>>(nodesb, W1a, W1b, Wqb, Wkb, bq, bk, U, V, Qm, Km);
  edge_kernel<<<Bsz * Nn * Nn / 8, 256, 0, stream>>>(U, V, be1, We2, be2, edge);
  final_kernel<<<Bsz, 256, 0, stream>>>(Qm, Km, edge, nodesf, Wout, bout, out);
}
